// MoE_18897856102780
// MI455X (gfx1250) — matrix operand order repaired and checked by reading, not compiled, not run on hardware
//
#include <hip/hip_runtime.h>
#include <hip/hip_bf16.h>

typedef __attribute__((ext_vector_type(8)))  float  v8f;
typedef __attribute__((ext_vector_type(16))) __bf16 v16bf;

struct Frag { uint4 lo, hi; };   // 32 bytes, bit-identical to v16bf

// Two fp32 -> packed bf16 (RNE) in one instruction: D[15:0]=bf16(lo), D[31:16]=bf16(hi)
__device__ __forceinline__ unsigned pack2bf(float lo, float hi) {
    unsigned r;
    asm("v_cvt_pk_bf16_f32 %0, %1, %2" : "=v"(r) : "v"(lo), "v"(hi));
    return r;
}

__device__ __forceinline__ v8f wmma_bf16(v16bf a, v16bf b, v8f c) {
    return __builtin_amdgcn_wmma_f32_16x16x32_bf16(
        /*neg_a=*/false, a, /*neg_b=*/false, b,
        /*c_mod=*/(short)0, c, /*reuse_a=*/false, /*reuse_b=*/false);
}

// ---------------------------------------------------------------------------
// Problem constants
// ---------------------------------------------------------------------------
#define Bsz 8192
#define DM  512
#define NE  4
#define Pdim 1024
#define Fdim 2048

// GEMM tiling
#define BM 128      // rows (tokens) per block
#define BP 32       // P columns per block
#define BK 32       // K per iteration
#define KP 40       // padded LDS K-stride in halves (80 bytes, 16B-aligned rows)

__device__ __forceinline__ const float* pick_mod(
    int mod, const float* x0, const float* x1, const float* x2, const float* x3)
{
    const float* p = x0;
    if (mod == 1) p = x1;
    if (mod == 2) p = x2;
    if (mod == 3) p = x3;
    return p;
}

// ---------------------------------------------------------------------------
// Kernel 1: gating  (logits -> softmax -> keep top-2 probs, zero the rest)
// ---------------------------------------------------------------------------
__global__ __launch_bounds__(128) void gate_kernel(
    const float* __restrict__ x0, const float* __restrict__ x1,
    const float* __restrict__ x2, const float* __restrict__ x3,
    const float* __restrict__ Wg, float* __restrict__ wgate)
{
    const int b   = blockIdx.x;
    const int tid = threadIdx.x;

    float a0 = 0.f, a1 = 0.f, a2 = 0.f, a3 = 0.f;
#pragma unroll
    for (int m = 0; m < 4; ++m) {
        const float* xp = pick_mod(m, x0, x1, x2, x3);
        for (int fl = tid; fl < DM; fl += 128) {
            const float xv = xp[(size_t)b * DM + fl];
            const int f = m * DM + fl;
            a0 += xv * Wg[0 * Fdim + f];
            a1 += xv * Wg[1 * Fdim + f];
            a2 += xv * Wg[2 * Fdim + f];
            a3 += xv * Wg[3 * Fdim + f];
        }
    }

    __shared__ float sm[4][128];
    sm[0][tid] = a0; sm[1][tid] = a1; sm[2][tid] = a2; sm[3][tid] = a3;
    __syncthreads();
    for (int off = 64; off > 0; off >>= 1) {
        if (tid < off) {
#pragma unroll
            for (int e = 0; e < 4; ++e) sm[e][tid] += sm[e][tid + off];
        }
        __syncthreads();
    }

    if (tid == 0) {
        float l[4] = { sm[0][0], sm[1][0], sm[2][0], sm[3][0] };
        float mx = fmaxf(fmaxf(l[0], l[1]), fmaxf(l[2], l[3]));
        float p[4]; float s = 0.f;
#pragma unroll
        for (int e = 0; e < 4; ++e) { p[e] = __expf(l[e] - mx); s += p[e]; }
        const float inv = 1.0f / s;
#pragma unroll
        for (int e = 0; e < 4; ++e) p[e] *= inv;
        // top-2 (lowest index wins ties, like lax.top_k)
        int i1 = 0;
#pragma unroll
        for (int e = 1; e < 4; ++e) if (p[e] > p[i1]) i1 = e;
        int i2 = -1;
#pragma unroll
        for (int e = 0; e < 4; ++e)
            if (e != i1 && (i2 < 0 || p[e] > p[i2])) i2 = e;
#pragma unroll
        for (int e = 0; e < 4; ++e)
            wgate[b * 4 + e] = (e == i1 || e == i2) ? p[e] : 0.0f;
    }
}

// ---------------------------------------------------------------------------
// Kernel 2: fused MoE GEMM.  Each block owns a (BM x BP) tile of preds and
// computes all 4 experts' partial GEMMs with shared A fragments, then does
// preds[b,p] = sum_e wgate[b,e] * (acc_e[b,p] + be[e,p]).  No atomics.
// Software-pipelined: tile k+1 global loads overlap tile k WMMAs.
// ---------------------------------------------------------------------------
__global__ __launch_bounds__(256) void moe_gemm_kernel(
    const float* __restrict__ x0, const float* __restrict__ x1,
    const float* __restrict__ x2, const float* __restrict__ x3,
    const float* __restrict__ We, const float* __restrict__ be,
    const float* __restrict__ wgate, float* __restrict__ preds)
{
    __shared__ __align__(16) unsigned short As[BM * KP];      // [row][k]
    __shared__ __align__(16) unsigned short Bs[NE * BP * KP]; // [e][p][k] (transposed)

    const int tid  = threadIdx.x;
    const int bM   = blockIdx.x * BM;   // token base
    const int pB   = blockIdx.y * BP;   // P base

    const int wave = tid >> 5;
    const int lane = tid & 31;
    const int wm   = wave & 3;          // M quadrant: rows wm*32 .. +31
    const int wp   = wave >> 2;         // P half:    cols wp*16 .. +15
    const int r    = lane & 15;
    const int hi   = lane >> 4;

    const int kq   = (tid & 7) * 4;     // A: k-quad / B: p-quad selector

    v8f acc[NE][2];
#pragma unroll
    for (int e = 0; e < NE; ++e)
#pragma unroll
        for (int s = 0; s < 2; ++s) acc[e][s] = (v8f){0,0,0,0,0,0,0,0};

    // register staging for the software pipeline
    float4 av[4];               // A: 4 rows x (4 k) each
    float4 bva[2], bvb[2];      // B: 2 tasks x (k2 row, k2+1 row) of a p-quad

    auto prefA = [&](const float* xp, int fbase) {
#pragma unroll
        for (int i = 0; i < 4; ++i) {
            const int row = i * 32 + (tid >> 3);
            av[i] = *(const float4*)(xp + (size_t)(bM + row) * DM + fbase + kq);
        }
    };
    auto prefB = [&](int k0) {
#pragma unroll
        for (int i = 0; i < 2; ++i) {
            const int tid2 = i * 256 + tid;
            const int rp   = tid2 >> 3;          // 0..63 : (e, k-pair)
            const int e    = rp >> 4;
            const int k2   = (rp & 15) * 2;
            const int pq   = (tid2 & 7) * 4;
            const float* src = We + ((size_t)e * Fdim + (k0 + k2)) * Pdim + pB + pq;
            bva[i] = *(const float4*)(src);          // row k2
            bvb[i] = *(const float4*)(src + Pdim);   // row k2+1
        }
    };
    auto stage = [&]() {
        // A: pack 4 bf16 per row into one b64 store (two v_cvt_pk_bf16_f32)
#pragma unroll
        for (int i = 0; i < 4; ++i) {
            const int row = i * 32 + (tid >> 3);
            uint2 pk;
            pk.x = pack2bf(av[i].x, av[i].y);
            pk.y = pack2bf(av[i].z, av[i].w);
            *(uint2*)(&As[row * KP + kq]) = pk;
        }
        // B: transpose; pack (k2, k2+1) bf16 pairs into b32 stores
#pragma unroll
        for (int i = 0; i < 2; ++i) {
            const int tid2 = i * 256 + tid;
            const int rp   = tid2 >> 3;
            const int e    = rp >> 4;
            const int k2   = (rp & 15) * 2;
            const int pq   = (tid2 & 7) * 4;
            unsigned* dst = (unsigned*)&Bs[(e * BP + pq) * KP + k2]; // KP/2=20 uints per row
            dst[0 * (KP / 2)] = pack2bf(bva[i].x, bvb[i].x);
            dst[1 * (KP / 2)] = pack2bf(bva[i].y, bvb[i].y);
            dst[2 * (KP / 2)] = pack2bf(bva[i].z, bvb[i].z);
            dst[3 * (KP / 2)] = pack2bf(bva[i].w, bvb[i].w);
        }
    };

    for (int mod = 0; mod < 4; ++mod) {
        const float* xp = pick_mod(mod, x0, x1, x2, x3);
        prefA(xp, 0);
        prefB(mod * DM);
        for (int kk = 0; kk < DM; kk += BK) {
            __syncthreads();            // previous tile's LDS reads complete
            stage();                    // regs -> LDS (bf16)
            __syncthreads();            // staging visible

            // prefetch next tile of this modality (overlaps with WMMAs below)
            if (kk + BK < DM) {
                prefA(xp, kk + BK);
                prefB(mod * DM + kk + BK);
            }

            // ---- fragments + WMMA ----
            v16bf afr[2];
#pragma unroll
            for (int s = 0; s < 2; ++s) {
                const unsigned short* ar = &As[(wm * 32 + s * 16 + r) * KP];
                Frag f;
                f.lo = *(const uint4*)(ar + 8 * hi);        // K = 8*hi .. +7
                f.hi = *(const uint4*)(ar + 16 + 8 * hi);   // K = 16+8*hi .. +7
                afr[s] = __builtin_bit_cast(v16bf, f);
            }
#pragma unroll
            for (int e = 0; e < NE; ++e) {
                const unsigned short* br = &Bs[(e * BP + wp * 16 + r) * KP + 8 * hi];
                Frag f;
                f.lo = *(const uint4*)(br);                 // K = 8*hi .. +7
                f.hi = *(const uint4*)(br + 16);            // K = 16+8*hi .. +7
                const v16bf bfr = __builtin_bit_cast(v16bf, f);
                acc[e][0] = wmma_bf16(afr[0], bfr, acc[e][0]);
                acc[e][1] = wmma_bf16(afr[1], bfr, acc[e][1]);
            }
        }
    }

    // ---- epilogue: weighted expert combine + bias ----
    const int pg = pB + wp * 16 + r;                        // global P column
    float bev[NE];
#pragma unroll
    for (int e = 0; e < NE; ++e) bev[e] = be[e * Pdim + pg];

#pragma unroll
    for (int s = 0; s < 2; ++s) {
#pragma unroll
        for (int i = 0; i < 8; ++i) {
            const int b = bM + wm * 32 + s * 16 + hi * 8 + i;  // C layout: VGPR i, lane-half hi
            const float4 w4 = *(const float4*)(wgate + b * 4);
            const float v = w4.x * (acc[0][s][i] + bev[0])
                          + w4.y * (acc[1][s][i] + bev[1])
                          + w4.z * (acc[2][s][i] + bev[2])
                          + w4.w * (acc[3][s][i] + bev[3]);
            preds[(size_t)b * Pdim + pg] = v;
        }
    }
}

// ---------------------------------------------------------------------------
// Kernels 3+4: deterministic MSE reduction
// ---------------------------------------------------------------------------
__global__ __launch_bounds__(256) void loss_partial_kernel(
    const float* __restrict__ preds, const float* __restrict__ label,
    float* __restrict__ partial)
{
    const int tid = threadIdx.x;
    const size_t base = (size_t)blockIdx.x * 1024 + tid * 4;
    float s = 0.f;
#pragma unroll
    for (int j = 0; j < 4; ++j) {
        const float d = preds[base + j] - label[base + j];  // preds is d_out+1: scalar loads
        s += d * d;
    }
    __shared__ float sm[256];
    sm[tid] = s;
    __syncthreads();
    for (int off = 128; off > 0; off >>= 1) {
        if (tid < off) sm[tid] += sm[tid + off];
        __syncthreads();
    }
    if (tid == 0) partial[blockIdx.x] = sm[0];
}

__global__ __launch_bounds__(256) void loss_final_kernel(
    const float* __restrict__ partial, float* __restrict__ out)
{
    const int tid = threadIdx.x;
    float s = 0.f;
    for (int i = tid; i < 8192; i += 256) s += partial[i];
    __shared__ float sm[256];
    sm[tid] = s;
    __syncthreads();
    for (int off = 128; off > 0; off >>= 1) {
        if (tid < off) sm[tid] += sm[tid + off];
        __syncthreads();
    }
    if (tid == 0) out[0] = sm[0] * (1.0f / ((float)Bsz * (float)Pdim));
}

// ---------------------------------------------------------------------------
extern "C" void kernel_launch(void* const* d_in, const int* in_sizes, int n_in,
                              void* d_out, int out_size, void* d_ws, size_t ws_size,
                              hipStream_t stream) {
    const float* x0    = (const float*)d_in[0];
    const float* x1    = (const float*)d_in[1];
    const float* x2    = (const float*)d_in[2];
    const float* x3    = (const float*)d_in[3];
    const float* label = (const float*)d_in[4];
    const float* Wg    = (const float*)d_in[5];
    const float* We    = (const float*)d_in[6];
    const float* be    = (const float*)d_in[7];
    (void)in_sizes; (void)n_in; (void)ws_size; (void)out_size;

    float* out   = (float*)d_out;       // out[0] = loss, out+1 = preds [B][P]
    float* preds = out + 1;

    float* wsf     = (float*)d_ws;
    float* wgate   = wsf;               // [B][4]          = 128 KB
    float* partial = wsf + Bsz * 4;     // [8192]          =  32 KB

    // 1) gating weights (softmax probs, top-2 kept, others zeroed)
    gate_kernel<<<dim3(Bsz), dim3(128), 0, stream>>>(x0, x1, x2, x3, Wg, wgate);

    // 2) fused 4-expert GEMM + weighted combine + bias -> preds
    moe_gemm_kernel<<<dim3(Bsz / BM, Pdim / BP), dim3(256), 0, stream>>>(
        x0, x1, x2, x3, We, be, wgate, preds);

    // 3) + 4) deterministic MSE
    loss_partial_kernel<<<dim3(8192), dim3(256), 0, stream>>>(preds, label, partial);
    loss_final_kernel<<<dim3(1), dim3(256), 0, stream>>>(partial, out);
}
